// AdaptiveSoftmax_83614423319285
// MI455X (gfx1250) — matrix operand order repaired and checked by reading, not compiled, not run on hardware
//
#include <hip/hip_runtime.h>
#include <hip/hip_bf16.h>

// ---------------------------------------------------------------------------
// Adaptive softmax NLL, fused bf16-WMMA implementation for gfx1250 (MI455X).
//   head:  x[4096,1024] @ [W_head;W_cluster]^T  -> streaming logsumexp + gathers
//   tail1: (x@P1^T)[4096,256] @ W1^T            -> lse on compacted rows
//   tail2: (x@P2^T)[4096,64]  @ W2^T            -> lse on compacted rows
// ---------------------------------------------------------------------------

typedef __attribute__((ext_vector_type(16))) __bf16 v16bf;
typedef __attribute__((ext_vector_type(8)))  __bf16 v8bf;
typedef __attribute__((ext_vector_type(8)))  float  v8f;

#define L2E   1.44269504088896340736f   // log2(e)
#define LN2   0.69314718055994530942f   // ln(2)

#define NROWS 4096
#define DIM   1024
#define C0    20000
#define C1    40000
#define C2    50257

// padded B-buffer row counts (multiples of 128, >= splits*chunks*128)
#define HEAD_BROWS 20480   // 8 splits * 20 chunks * 128
#define T1_BROWS   20480   // 8 splits * 20 chunks * 128
#define T2_BROWS   11264   // 8 splits * 11 chunks * 128
#define NSPLITS    8

// ---------------------------------------------------------------------------
// helpers
// ---------------------------------------------------------------------------
__device__ __forceinline__ v16bf make_frag(const __bf16* p0, const __bf16* p1) {
  v8bf lo = *(const v8bf*)p0;
  v8bf hi = *(const v8bf*)p1;
  return __builtin_shufflevector(lo, hi, 0,1,2,3,4,5,6,7,8,9,10,11,12,13,14,15);
}

// ---------------------------------------------------------------------------
// init: zero scalar output and compaction counters
// ---------------------------------------------------------------------------
__global__ void init_kernel(float* out, int* cnt1, int* cnt2) {
  if (threadIdx.x == 0) { out[0] = 0.f; *cnt1 = 0; *cnt2 = 0; }
}

// ---------------------------------------------------------------------------
// fp32 -> bf16 convert with zero row padding
// ---------------------------------------------------------------------------
__global__ void convert_pad_kernel(const float* __restrict__ src, __bf16* __restrict__ dst,
                                   int src_rows, int dst_rows, int cols) {
  long long total = (long long)dst_rows * cols;
  long long i = (long long)blockIdx.x * blockDim.x + threadIdx.x;
  if (i >= total) return;
  int r = (int)(i / cols);
  float v = (r < src_rows) ? src[i] : 0.f;
  dst[i] = (__bf16)v;
}

// ---------------------------------------------------------------------------
// target -> head target column (or -1) + compacted row lists for the tails
// ---------------------------------------------------------------------------
__global__ void compact_kernel(const int* __restrict__ target,
                               int* __restrict__ tcolh,
                               int* __restrict__ list1, int* __restrict__ tcol1, int* cnt1,
                               int* __restrict__ list2, int* __restrict__ tcol2, int* cnt2) {
  int r = blockIdx.x * blockDim.x + threadIdx.x;
  if (r >= NROWS) return;
  int t = target[r];
  tcolh[r] = (t < C0) ? t : -1;
  if (t >= C0 && t < C1) {
    int p = atomicAdd(cnt1, 1);
    list1[p] = r; tcol1[p] = t - C0;
  } else if (t >= C1) {
    int p = atomicAdd(cnt2, 1);
    list2[p] = r; tcol2[p] = t - C1;
  }
}

// ---------------------------------------------------------------------------
// fused GEMM (bf16 WMMA, f32 accum) + flat logsumexp + column gathers.
// block = 256 threads (8 waves). Wave w owns rows m0..m0+15 of a 128-row
// M-block; grid.y = N-split. Per 128-col chunk: stage 128x32 bf16 B tile in
// LDS each K-step, 8x v_wmma_f32_16x16x32_bf16 per wave per K-step.
// ---------------------------------------------------------------------------
__global__ void __launch_bounds__(256)
gemm_lse_kernel(const __bf16* __restrict__ A, int K,
                const __bf16* __restrict__ B,
                const int* __restrict__ rowlist,   // nullptr => identity rows
                const int* __restrict__ cnt_ptr,   // nullptr => fixed_cnt
                int fixed_cnt,
                const int* __restrict__ tcol,      // per logical row: target col or -1
                int chunks_per_split, int n_valid,
                int sp0_col, int sp1_col,
                float* __restrict__ parts, int nsplits,
                float* __restrict__ tgt_out,
                float* __restrict__ sp0_out, float* __restrict__ sp1_out) {
  int cnt = cnt_ptr ? *cnt_ptr : fixed_cnt;
  int mBase = blockIdx.x * 128;
  if (mBase >= cnt) return;

  int wave = threadIdx.x >> 5;
  int lane = threadIdx.x & 31;
  int m0   = mBase + wave * 16;
  int split = blockIdx.y;

  __shared__ __align__(16) __bf16 Bt[128 * 32];

  // A fragment row for this lane (same row for lane and lane+16 halves)
  int aLogical = m0 + (lane & 15);
  int aRow = aLogical;
  if (rowlist) aRow = (aLogical < cnt) ? rowlist[aLogical] : 0;
  const __bf16* Arow = A + (size_t)aRow * K;
  int aOff0 = (lane >> 4) << 3;            // K sub-offset: 0 or 8

  // accumulator rows held by this lane (C/D layout: lanes 0-15 -> M 0..7)
  int rBase = m0 + ((lane >> 4) << 3);
  int tc[8];
#pragma unroll
  for (int v = 0; v < 8; ++v) {
    int ri = rBase + v;
    tc[v] = (tcol && ri < cnt) ? tcol[ri] : -1;
  }

  float srun[8] = {0.f,0.f,0.f,0.f,0.f,0.f,0.f,0.f};
  const v8f zero8 = {0.f,0.f,0.f,0.f,0.f,0.f,0.f,0.f};

  int ksteps  = K >> 5;
  int chunk0  = split * chunks_per_split;
  int ldCol   = threadIdx.x >> 1;          // 0..127: B tile column to stage
  int ldHalf  = (threadIdx.x & 1) << 4;    // 0 or 16 (K half)
  int bKHalf  = (lane >> 4) << 3;          // 0 or 8: B fragment K sub-offset

  for (int c = 0; c < chunks_per_split; ++c) {
    int nBase = (chunk0 + c) << 7;
    const __bf16* Bbase = B + (size_t)nBase * K;
    v8f acc[8];
#pragma unroll
    for (int t = 0; t < 8; ++t) acc[t] = zero8;

    for (int ks = 0; ks < ksteps; ++ks) {
      int k0 = ks << 5;
      // cooperative B tile load: col-major [col][32 k] in LDS
      {
        const __bf16* src = Bbase + (size_t)ldCol * K + k0 + ldHalf;
        *(v8bf*)&Bt[ldCol * 32 + ldHalf]     = *(const v8bf*)src;
        *(v8bf*)&Bt[ldCol * 32 + ldHalf + 8] = *(const v8bf*)(src + 8);
        if (ks + 1 < ksteps) __builtin_prefetch(src + 32, 0, 1);
      }
      __syncthreads();

      v16bf afrag = make_frag(Arow + k0 + aOff0, Arow + k0 + 16 + aOff0);
      const __bf16* bl = &Bt[(lane & 15) * 32 + bKHalf];
#pragma unroll
      for (int t = 0; t < 8; ++t) {
        v16bf bfrag = make_frag(bl + t * (16 * 32), bl + t * (16 * 32) + 16);
        acc[t] = __builtin_amdgcn_wmma_f32_16x16x32_bf16(
            false, afrag, false, bfrag, (short)0, acc[t], false, false);
      }
      __syncthreads();
    }

    // epilogue: flat sum of exp(logit) + gathers (logits bounded ~|5|)
    int col0 = nBase + (lane & 15);
#pragma unroll
    for (int t = 0; t < 8; ++t) {
      int col = col0 + t * 16;
      bool valid = col < n_valid;
      if (valid) {
#pragma unroll
        for (int v = 0; v < 8; ++v) {
          float xl = acc[t][v];
          srun[v] += exp2f(xl * L2E);
          if (col == tc[v]) tgt_out[rBase + v] = xl;
        }
        if (col == sp0_col) {
#pragma unroll
          for (int v = 0; v < 8; ++v) sp0_out[rBase + v] = acc[t][v];
        }
        if (col == sp1_col) {
#pragma unroll
          for (int v = 0; v < 8; ++v) sp1_out[rBase + v] = acc[t][v];
        }
      }
    }
  }

  // reduce partial sums across the 16 lanes sharing each row group
#pragma unroll
  for (int v = 0; v < 8; ++v) {
    float s = srun[v];
    s += __shfl_xor(s, 1, 32);
    s += __shfl_xor(s, 2, 32);
    s += __shfl_xor(s, 4, 32);
    s += __shfl_xor(s, 8, 32);
    srun[v] = s;
  }
  if ((lane & 15) == 0) {
#pragma unroll
    for (int v = 0; v < 8; ++v) {
      int ri = rBase + v;
      if (ri < cnt) parts[(size_t)ri * nsplits + split] = srun[v];
    }
  }
}

// ---------------------------------------------------------------------------
// plain GEMM (bf16 WMMA) storing bf16 output: h = x @ P^T  (projections)
// identity rows, single N split.
// ---------------------------------------------------------------------------
__global__ void __launch_bounds__(256)
gemm_store_kernel(const __bf16* __restrict__ A, int K,
                  const __bf16* __restrict__ B,
                  int n_chunks, int n_out, __bf16* __restrict__ H) {
  int wave = threadIdx.x >> 5;
  int lane = threadIdx.x & 31;
  int m0 = blockIdx.x * 128 + wave * 16;

  __shared__ __align__(16) __bf16 Bt[128 * 32];

  const __bf16* Arow = A + (size_t)(m0 + (lane & 15)) * K;
  int aOff0  = (lane >> 4) << 3;
  int rBase  = m0 + ((lane >> 4) << 3);
  int ldCol  = threadIdx.x >> 1;
  int ldHalf = (threadIdx.x & 1) << 4;
  int bKHalf = (lane >> 4) << 3;
  int ksteps = K >> 5;
  const v8f zero8 = {0.f,0.f,0.f,0.f,0.f,0.f,0.f,0.f};

  for (int c = 0; c < n_chunks; ++c) {
    int nBase = c << 7;
    const __bf16* Bbase = B + (size_t)nBase * K;
    v8f acc[8];
#pragma unroll
    for (int t = 0; t < 8; ++t) acc[t] = zero8;

    for (int ks = 0; ks < ksteps; ++ks) {
      int k0 = ks << 5;
      {
        const __bf16* src = Bbase + (size_t)ldCol * K + k0 + ldHalf;
        *(v8bf*)&Bt[ldCol * 32 + ldHalf]     = *(const v8bf*)src;
        *(v8bf*)&Bt[ldCol * 32 + ldHalf + 8] = *(const v8bf*)(src + 8);
      }
      __syncthreads();
      v16bf afrag = make_frag(Arow + k0 + aOff0, Arow + k0 + 16 + aOff0);
      const __bf16* bl = &Bt[(lane & 15) * 32 + bKHalf];
#pragma unroll
      for (int t = 0; t < 8; ++t) {
        v16bf bfrag = make_frag(bl + t * (16 * 32), bl + t * (16 * 32) + 16);
        acc[t] = __builtin_amdgcn_wmma_f32_16x16x32_bf16(
            false, afrag, false, bfrag, (short)0, acc[t], false, false);
      }
      __syncthreads();
    }

    int col0 = nBase + (lane & 15);
#pragma unroll
    for (int t = 0; t < 8; ++t) {
      int col = col0 + t * 16;
      if (col < n_out) {
#pragma unroll
        for (int v = 0; v < 8; ++v)
          H[(size_t)(rBase + v) * n_out + col] = (__bf16)acc[t][v];
      }
    }
  }
}

// ---------------------------------------------------------------------------
// head reduce: lse per row, cluster-0 NLL contribution, hc1/hc2 priors
// ---------------------------------------------------------------------------
__global__ void head_reduce_kernel(const float* __restrict__ parts, int nsplits,
                                   const int* __restrict__ tcolh,
                                   const float* __restrict__ tgt0,
                                   const float* __restrict__ sp0,
                                   const float* __restrict__ sp1,
                                   float* __restrict__ hc1, float* __restrict__ hc2,
                                   float* __restrict__ out) {
  int r = blockIdx.x * blockDim.x + threadIdx.x;
  float contrib = 0.f;
  if (r < NROWS) {
    float s = 0.f;
    for (int i = 0; i < nsplits; ++i) s += parts[(size_t)r * nsplits + i];
    float lse = LN2 * log2f(s);
    hc1[r] = sp0[r] - lse;
    hc2[r] = sp1[r] - lse;
    if (tcolh[r] >= 0) contrib = -(tgt0[r] - lse);
  }
  __shared__ float red[256];
  red[threadIdx.x] = contrib;
  __syncthreads();
  for (int o = 128; o > 0; o >>= 1) {
    if (threadIdx.x < o) red[threadIdx.x] += red[threadIdx.x + o];
    __syncthreads();
  }
  if (threadIdx.x == 0) atomicAdd(out, red[0]);
}

// ---------------------------------------------------------------------------
// tail reduce over compacted rows: -(hc[row] + logit_t - lse)
// ---------------------------------------------------------------------------
__global__ void tail_reduce_kernel(const float* __restrict__ parts, int nsplits,
                                   const float* __restrict__ tgt,
                                   const int* __restrict__ list,
                                   const int* __restrict__ cnt_ptr,
                                   const float* __restrict__ hc,
                                   float* __restrict__ out) {
  int i = blockIdx.x * blockDim.x + threadIdx.x;
  int cnt = *cnt_ptr;
  float contrib = 0.f;
  if (i < cnt) {
    float s = 0.f;
    for (int k = 0; k < nsplits; ++k) s += parts[(size_t)i * nsplits + k];
    float lse = LN2 * log2f(s);
    contrib = -(hc[list[i]] + tgt[i] - lse);
  }
  __shared__ float red[256];
  red[threadIdx.x] = contrib;
  __syncthreads();
  for (int o = 128; o > 0; o >>= 1) {
    if (threadIdx.x < o) red[threadIdx.x] += red[threadIdx.x + o];
    __syncthreads();
  }
  if (threadIdx.x == 0) atomicAdd(out, red[0]);
}

// ---------------------------------------------------------------------------
// host launcher
// ---------------------------------------------------------------------------
static inline size_t alignUp(size_t v) { return (v + 255) & ~(size_t)255; }

extern "C" void kernel_launch(void* const* d_in, const int* in_sizes, int n_in,
                              void* d_out, int out_size, void* d_ws, size_t ws_size,
                              hipStream_t stream) {
  const float* x         = (const float*)d_in[0];
  const int*   target    = (const int*)d_in[1];
  const float* W_head    = (const float*)d_in[2];
  const float* W_cluster = (const float*)d_in[3];
  const float* P1        = (const float*)d_in[4];
  const float* W1        = (const float*)d_in[5];
  const float* P2        = (const float*)d_in[6];
  const float* W2        = (const float*)d_in[7];
  float* out = (float*)d_out;

  // ---- workspace carving -------------------------------------------------
  char* w = (char*)d_ws;
  size_t off = 0;
  auto carve = [&](size_t bytes) -> char* {
    char* p = w + off;
    off = alignUp(off + bytes);
    return p;
  };
  __bf16* xbf  = (__bf16*)carve((size_t)NROWS * DIM * 2);
  __bf16* Bh   = (__bf16*)carve((size_t)HEAD_BROWS * DIM * 2);
  __bf16* B1   = (__bf16*)carve((size_t)T1_BROWS * 256 * 2);
  __bf16* B2   = (__bf16*)carve((size_t)T2_BROWS * 64 * 2);
  __bf16* P1b  = (__bf16*)carve((size_t)256 * DIM * 2);
  __bf16* P2b  = (__bf16*)carve((size_t)128 * DIM * 2);
  __bf16* h1   = (__bf16*)carve((size_t)NROWS * 256 * 2);
  __bf16* h2   = (__bf16*)carve((size_t)NROWS * 64 * 2);
  float* partsH = (float*)carve((size_t)NROWS * NSPLITS * 4);
  float* parts1 = (float*)carve((size_t)NROWS * NSPLITS * 4);
  float* parts2 = (float*)carve((size_t)NROWS * NSPLITS * 4);
  float* tgt0   = (float*)carve((size_t)NROWS * 4);
  float* sp0    = (float*)carve((size_t)NROWS * 4);
  float* sp1    = (float*)carve((size_t)NROWS * 4);
  float* hc1    = (float*)carve((size_t)NROWS * 4);
  float* hc2    = (float*)carve((size_t)NROWS * 4);
  float* tgt1   = (float*)carve((size_t)NROWS * 4);
  float* tgt2   = (float*)carve((size_t)NROWS * 4);
  int* tcolh = (int*)carve((size_t)NROWS * 4);
  int* list1 = (int*)carve((size_t)NROWS * 4);
  int* tcol1 = (int*)carve((size_t)NROWS * 4);
  int* list2 = (int*)carve((size_t)NROWS * 4);
  int* tcol2 = (int*)carve((size_t)NROWS * 4);
  int* cnt1  = (int*)carve(64);
  int* cnt2  = (int*)carve(64);
  (void)ws_size; (void)in_sizes; (void)n_in; (void)out_size;

  // ---- init + compaction -------------------------------------------------
  init_kernel<<<1, 64, 0, stream>>>(out, cnt1, cnt2);
  compact_kernel<<<(NROWS + 255) / 256, 256, 0, stream>>>(
      target, tcolh, list1, tcol1, cnt1, list2, tcol2, cnt2);

  // ---- bf16 conversions with zero padding --------------------------------
  auto conv = [&](const float* src, __bf16* dst, int sr, int dr, int cols) {
    long long total = (long long)dr * cols;
    int blocks = (int)((total + 255) / 256);
    convert_pad_kernel<<<blocks, 256, 0, stream>>>(src, dst, sr, dr, cols);
  };
  conv(x, xbf, NROWS, NROWS, DIM);
  conv(W_head,    Bh,                         C0, C0,              DIM); // rows 0..19999
  conv(W_cluster, Bh + (size_t)C0 * DIM,       2, HEAD_BROWS - C0, DIM); // rows 20000..20001 + zero pad
  conv(W1, B1, C1 - C0, T1_BROWS, 256);
  conv(W2, B2, C2 - C1, T2_BROWS, 64);
  conv(P1, P1b, 256, 256, DIM);
  conv(P2, P2b, 64, 128, DIM);

  // ---- projections: h1 = x@P1^T (bf16), h2 = x@P2^T -----------------------
  gemm_store_kernel<<<dim3(NROWS / 128, 1), 256, 0, stream>>>(xbf, DIM, P1b, 2, 256, h1);
  gemm_store_kernel<<<dim3(NROWS / 128, 1), 256, 0, stream>>>(xbf, DIM, P2b, 1, 64, h2);

  // ---- head fused GEMM + lse: 20480 padded cols, 20002 valid --------------
  gemm_lse_kernel<<<dim3(NROWS / 128, NSPLITS), 256, 0, stream>>>(
      xbf, DIM, Bh, nullptr, nullptr, NROWS, tcolh,
      /*chunks_per_split=*/20, /*n_valid=*/C0 + 2,
      /*sp0_col=*/C0, /*sp1_col=*/C0 + 1,
      partsH, NSPLITS, tgt0, sp0, sp1);

  // ---- tail1 fused GEMM + lse on compacted rows ---------------------------
  gemm_lse_kernel<<<dim3(NROWS / 128, NSPLITS), 256, 0, stream>>>(
      h1, 256, B1, list1, cnt1, 0, tcol1,
      /*chunks_per_split=*/20, /*n_valid=*/C1 - C0,
      -1, -1, parts1, NSPLITS, tgt1, nullptr, nullptr);

  // ---- tail2 fused GEMM + lse on compacted rows ---------------------------
  gemm_lse_kernel<<<dim3(NROWS / 128, NSPLITS), 256, 0, stream>>>(
      h2, 64, B2, list2, cnt2, 0, tcol2,
      /*chunks_per_split=*/11, /*n_valid=*/C2 - C1,
      -1, -1, parts2, NSPLITS, tgt2, nullptr, nullptr);

  // ---- reductions ---------------------------------------------------------
  head_reduce_kernel<<<NROWS / 256, 256, 0, stream>>>(
      partsH, NSPLITS, tcolh, tgt0, sp0, sp1, hc1, hc2, out);
  tail_reduce_kernel<<<NROWS / 256, 256, 0, stream>>>(
      parts1, NSPLITS, tgt1, list1, cnt1, hc1, out);
  tail_reduce_kernel<<<NROWS / 256, 256, 0, stream>>>(
      parts2, NSPLITS, tgt2, list2, cnt2, hc2, out);
}
